// LSTM_76630806496011
// MI455X (gfx1250) — matrix operand order repaired and checked by reading, not compiled, not run on hardware
//
#include <hip/hip_runtime.h>

// ---------------------------------------------------------------------------
// Fused LSTM cell for MI455X (gfx1250, wave32, WMMA + TDM), two-pass:
//   pass 1: convert x, h0, w_ih, w_hh f32 -> bf16 into d_ws (once).
//   pass 2: gates = [x|h] @ [w_ih|w_hh]^T via v_wmma_f32_16x16x32_bf16 with
//           Tensor-Data-Mover staging into double-buffered LDS, fused
//           bias + sigmoid/tanh epilogue writing h_new, c_new.
// Tile: workgroup = 256 rows x 32 cols x 4 gates; 8 waves, each wave owns
// 32 rows (2 m-tiles) so every B fragment feeds 2 WMMAs.
// ---------------------------------------------------------------------------

typedef __bf16        bf16x16 __attribute__((ext_vector_type(16)));
typedef float         v8f     __attribute__((ext_vector_type(8)));
typedef float         f32x4   __attribute__((ext_vector_type(4)));
typedef unsigned int  u32x4   __attribute__((ext_vector_type(4)));
typedef unsigned int  u32x2   __attribute__((ext_vector_type(2)));
typedef int           i32x4   __attribute__((ext_vector_type(4)));
typedef int           i32x8   __attribute__((ext_vector_type(8)));

#define B_ROWS 4096
#define DIM    1024
#define HID    1024
#define KTOT   2048
#define NELEM  (4u*1024u*1024u)

#define BM     256          // workgroup tile rows (8 waves x 32)
#define BN     32           // workgroup tile cols (per gate)
#define BK     64           // K per LDS stage
#define LDA    72           // LDS row stride in bf16 (64 + 8 pad)

#define A_SH   (BM * LDA)          // 18432 shorts
#define W_SH   (4 * BN * LDA)      //  9216 shorts
#define BUF_SH (A_SH + W_SH)       // 27648 shorts per buffer
#define SMEM_BYTES (2 * BUF_SH * 2) // 110592 B, double buffered

// ---- TDM availability / arity discrimination --------------------------------
#if defined(__has_builtin)
#  if __has_builtin(__builtin_amdgcn_tensor_load_to_lds)
#    define GPU_HAS_TDM 1
#  else
#    define GPU_HAS_TDM 0
#  endif
#else
#  define GPU_HAS_TDM 0
#endif

#if defined(__has_include)
#  if __has_include(<hip/amd_detail/amd_gfx1250_TDM.h>)
#    define TDM_SIX_ARGS 1     // therock-10.0 headers => 6-arg builtin
#  else
#    define TDM_SIX_ARGS 0     // ROCm 7.2 => 5-arg builtin
#  endif
#else
#  define TDM_SIX_ARGS 0
#endif

__device__ __forceinline__ unsigned int f2bf(float f) {
  unsigned int u = __float_as_uint(f);
  u += 0x7FFFu + ((u >> 16) & 1u);   // round-to-nearest-even
  return u >> 16;
}
__device__ __forceinline__ float sigmoidf_fast(float x) {
  return 1.0f / (1.0f + __expf(-x));
}
__device__ __forceinline__ float tanhf_fast(float x) {
  return 1.0f - 2.0f / (__expf(2.0f * x) + 1.0f);
}

#if GPU_HAS_TDM
__device__ __forceinline__ void wait_tdm() {
#if __has_builtin(__builtin_amdgcn_s_wait_tensorcnt)
  __builtin_amdgcn_s_wait_tensorcnt(0);
#else
  asm volatile("s_wait_tensorcnt 0x0" ::: "memory");
#endif
}

// Issue one TDM tile load: up to 3D tile (d2 = outer, stride1 elems between
// planes), data_size = 2B, LDS padding of 4 DWORDs every 32 DWORDs (row = 64
// bf16 = 128B) reproducing the LDA=72 layout. Tensor dims == tile dims (tile
// is in-bounds by construction).  D# layout per CDNA5 ISA section 8.
__device__ __forceinline__ void tdm_load_tile(unsigned ldsAddr, const void* gaddr,
                                              unsigned tileD0, unsigned tileD1,
                                              unsigned tileD2,
                                              unsigned long long stride0,
                                              unsigned long long stride1)
{
  const unsigned long long ga = (unsigned long long)gaddr;
  u32x4 g0;
  g0.x = 1u;                                               // count=1, user D#
  g0.y = ldsAddr;                                          // lds byte address
  g0.z = (unsigned)ga;                                     // global_addr lo
  g0.w = (unsigned)((ga >> 32) & 0x01FFFFFFull) | (2u << 30); // addr hi | type=2
  i32x8 g1;
  g1[0] = (int)((1u << 16)            // data_size = 2 bytes
              | (1u << 20)            // pad_enable
              | (4u << 22)            // pad_interval: 32 DWORDs (128 B row)
              | (3u << 25));          // pad_amount: 4 DWORDs (16 B)
  g1[1] = (int)((tileD0 & 0xFFFFu) << 16);                              // tensor_dim0 lo
  g1[2] = (int)(((tileD0 >> 16) & 0xFFFFu) | ((tileD1 & 0xFFFFu) << 16)); // d0 hi | tensor_dim1 lo
  g1[3] = (int)(((tileD1 >> 16) & 0xFFFFu) | ((tileD0 & 0xFFFFu) << 16)); // d1 hi | tile_dim0
  g1[4] = (int)((tileD1 & 0xFFFFu) | ((tileD2 & 0xFFFFu) << 16));        // tile_dim1 | tile_dim2
  g1[5] = (int)(stride0 & 0xFFFFFFFFull);                                // dim0_stride lo
  g1[6] = (int)(((stride0 >> 32) & 0xFFFFull) | ((stride1 & 0xFFFFull) << 16)); // |dim1_stride lo
  g1[7] = (int)((stride1 >> 16) & 0xFFFFFFFFull);                        // dim1_stride hi
  i32x4 g2;
  g2[0] = (int)tileD2;   // tensor_dim2 (0 for 2D tiles)
  g2[1] = 0; g2[2] = 0; g2[3] = 0;
  i32x4 g3 = {0, 0, 0, 0};
#if TDM_SIX_ARGS
  i32x8 g4 = {0, 0, 0, 0, 0, 0, 0, 0};
  __builtin_amdgcn_tensor_load_to_lds(g0, g1, g2, g3, g4, 0);
#else
  __builtin_amdgcn_tensor_load_to_lds(g0, g1, g2, g3, 0);
#endif
}
#endif // GPU_HAS_TDM

// ---- shared WMMA stage: 16 steps (2 x ks) x (4g x 2t), MT m-tiles ----------
// B fragments ping-pong between two live register buffers (bE/bO) so the
// allocator cannot coalesce them: step s+2's loads issue before step s's
// WMMAs, keeping ~4 b128 LDS loads in flight instead of wait-dscnt-0 stalls.
template<int MT>
__device__ __forceinline__ void wmma_stage(const unsigned short* __restrict__ sA,
                                           const unsigned short* __restrict__ sW,
                                           v8f (&acc)[4][2][MT],
                                           int wave, int nlo, int half)
{
  union Frag { u32x4 q[2]; bf16x16 v; };
  auto loadA = [&](int ks, int m) {
    Frag f;
    const int arow = wave * (16 * MT) + m * 16 + nlo;
    f.q[0] = *(const u32x4*)&sA[arow * LDA + ks + half * 8];
    f.q[1] = *(const u32x4*)&sA[arow * LDA + ks + 16 + half * 8];
    return f;
  };
  auto loadB = [&](int s) {
    const int ks = (s >> 3) * 32, g = (s >> 1) & 3, t = s & 1;
    const int brow = g * BN + t * 16 + nlo;
    Frag f;
    f.q[0] = *(const u32x4*)&sW[brow * LDA + ks + half * 8];
    f.q[1] = *(const u32x4*)&sW[brow * LDA + ks + 16 + half * 8];
    return f;
  };

  Frag a0[MT], a1[MT];
  #pragma unroll
  for (int m = 0; m < MT; ++m) a0[m] = loadA(0, m);
  Frag bE = loadB(0);       // even-step fragment
  Frag bO = loadB(1);       // odd-step fragment

  #pragma unroll
  for (int p = 0; p < 8; ++p) {
    const int s0 = 2 * p, s1 = 2 * p + 1;
    if (s0 == 4) {                       // prefetch ks=32 A fragments early
      #pragma unroll
      for (int m = 0; m < MT; ++m) a1[m] = loadA(32, m);
    }
    Frag bE2, bO2;
    if (s0 + 2 < 16) bE2 = loadB(s0 + 2);   // issue before s0's WMMAs
    {
      const int g = (s0 >> 1) & 3, t = s0 & 1;
      #pragma unroll
      for (int m = 0; m < MT; ++m) {
        const Frag& a = (s0 < 8) ? a0[m] : a1[m];
        acc[g][t][m] = __builtin_amdgcn_wmma_f32_16x16x32_bf16(
            false, a.v, false, bE.v, (short)0, acc[g][t][m], false, false);
      }
    }
    if (s1 + 2 < 16) bO2 = loadB(s1 + 2);   // issue before s1's WMMAs
    {
      const int g = (s1 >> 1) & 3, t = s1 & 1;
      #pragma unroll
      for (int m = 0; m < MT; ++m) {
        const Frag& a = (s1 < 8) ? a0[m] : a1[m];
        acc[g][t][m] = __builtin_amdgcn_wmma_f32_16x16x32_bf16(
            false, a.v, false, bO.v, (short)0, acc[g][t][m], false, false);
      }
    }
    if (s0 + 2 < 16) bE = bE2;
    if (s1 + 2 < 16) bO = bO2;
  }
}

// ---- shared fused LSTM epilogue --------------------------------------------
template<int MT>
__device__ __forceinline__ void lstm_epilogue(v8f (&acc)[4][2][MT],
                                              const float* __restrict__ c0,
                                              const float* __restrict__ b_ih,
                                              const float* __restrict__ b_hh,
                                              float* __restrict__ out,
                                              int rowBase, int nBase,
                                              int wave, int nlo, int half)
{
  float bi[4][2];
  #pragma unroll
  for (int g = 0; g < 4; ++g)
    #pragma unroll
    for (int t = 0; t < 2; ++t) {
      const int col = g * HID + nBase + t * 16 + nlo;
      bi[g][t] = b_ih[col] + b_hh[col];
    }
  // C/D layout: VGPR r of lane L holds M = r + 8*(L>>4), N = L&15
  #pragma unroll
  for (int m = 0; m < MT; ++m) {
    #pragma unroll
    for (int r = 0; r < 8; ++r) {
      const int row = rowBase + wave * (16 * MT) + m * 16 + half * 8 + r;
      #pragma unroll
      for (int t = 0; t < 2; ++t) {
        const int col = nBase + t * 16 + nlo;
        const float gi = acc[0][t][m][r] + bi[0][t];
        const float gf = acc[1][t][m][r] + bi[1][t];
        const float gg = acc[2][t][m][r] + bi[2][t];
        const float go = acc[3][t][m][r] + bi[3][t];
        const float cp = c0[(size_t)row * HID + col];
        const float cn = sigmoidf_fast(gf) * cp + sigmoidf_fast(gi) * tanhf_fast(gg);
        const float hn = sigmoidf_fast(go) * tanhf_fast(cn);
        out[(size_t)row * HID + col] = hn;
        out[(size_t)B_ROWS * HID + (size_t)row * HID + col] = cn;
      }
    }
  }
}

// ---------------- pass 1: f32 -> bf16 bulk convert --------------------------
__global__ __launch_bounds__(256)
void convert_to_bf16(const float* __restrict__ s0, const float* __restrict__ s1,
                     const float* __restrict__ s2, const float* __restrict__ s3,
                     unsigned short* __restrict__ dst)
{
  const int which = blockIdx.y;
  const float* s = (which == 0) ? s0 : (which == 1) ? s1 : (which == 2) ? s2 : s3;
  const size_t base = (size_t)which * NELEM;
  const size_t i = ((size_t)blockIdx.x * 256 + threadIdx.x) * 8;
  f32x4 a = *(const f32x4*)(s + i);
  f32x4 b = *(const f32x4*)(s + i + 4);
  u32x4 p;
  p.x = f2bf(a.x) | (f2bf(a.y) << 16);
  p.y = f2bf(a.z) | (f2bf(a.w) << 16);
  p.z = f2bf(b.x) | (f2bf(b.y) << 16);
  p.w = f2bf(b.z) | (f2bf(b.w) << 16);
  *(u32x4*)(dst + base + i) = p;
}

// ---------------- pass 2: WMMA GEMM + fused epilogue ------------------------
__global__ __launch_bounds__(256)
void lstm_gemm_bf16(const unsigned short* __restrict__ xb,
                    const unsigned short* __restrict__ hb,
                    const unsigned short* __restrict__ wihb,
                    const unsigned short* __restrict__ whhb,
                    const float* __restrict__ c0,
                    const float* __restrict__ b_ih,
                    const float* __restrict__ b_hh,
                    float* __restrict__ out)
{
  extern __shared__ __align__(16) unsigned short smem[];   // 2 x (A | W) buffers

  const int tid  = threadIdx.x;
  const int wave = tid >> 5;
  const int lane = tid & 31;
  const int nlo  = lane & 15;
  const int half = lane >> 4;
  const int rowBase = blockIdx.y * BM;
  const int nBase   = blockIdx.x * BN;

  v8f acc[4][2][2];
  {
    v8f z = {};
    #pragma unroll
    for (int g = 0; g < 4; ++g)
      #pragma unroll
      for (int t = 0; t < 2; ++t)
        #pragma unroll
        for (int m = 0; m < 2; ++m) acc[g][t][m] = z;
  }

#if GPU_HAS_TDM
  // --- Tensor Data Mover staging: 2 descriptors/stage (A 2D, W as 3D tile
  // with tile_dim2 = 4 gates, plane stride = H*D). Double-buffered LDS,
  // one barrier per stage; TDM for stage s+1 streams while WMMAs run on s.
  auto issue_stage = [&](int kb, int idx) {
    const unsigned short* aS = (kb < DIM) ? xb   : hb;
    const unsigned short* wS = (kb < DIM) ? wihb : whhb;
    const int kOff = kb & (DIM - 1);
    const unsigned aL = (unsigned)(size_t)(smem + idx * BUF_SH);
    const unsigned wL = (unsigned)(size_t)(smem + idx * BUF_SH + A_SH);
    tdm_load_tile(aL, aS + (size_t)rowBase * DIM + kOff, BK, BM, 0,
                  (unsigned long long)DIM, 0ull);
    tdm_load_tile(wL, wS + (size_t)nBase * DIM + kOff, BK, BN, 4,
                  (unsigned long long)DIM, (unsigned long long)HID * DIM);
  };

  if (wave == 0) issue_stage(0, 0);
  for (int kb = 0; kb < KTOT; kb += BK) {
    const int idx = (kb >> 6) & 1;
    if (wave == 0) wait_tdm();                 // stage idx landed in LDS
    __syncthreads();                           // visible to all; buf idx^1 free
    if (wave == 0 && kb + BK < KTOT) issue_stage(kb + BK, idx ^ 1);
    wmma_stage<2>(smem + idx * BUF_SH, smem + idx * BUF_SH + A_SH,
                  acc, wave, nlo, half);
  }
#else
  // --- fallback staging: register-pipelined global->LDS copy, double buffer
  u32x4 pA[8], pW[4];
  auto load_regs = [&](int kb) {
    const unsigned short* aS = (kb < DIM) ? xb   : hb;
    const unsigned short* wS = (kb < DIM) ? wihb : whhb;
    const int kOff = kb & (DIM - 1);
    #pragma unroll
    for (int i = 0; i < 8; ++i) {
      const int u = tid + i * 256, row = u >> 3, k8 = u & 7;
      pA[i] = *(const u32x4*)(aS + (size_t)(rowBase + row) * DIM + kOff + k8 * 8);
    }
    #pragma unroll
    for (int i = 0; i < 4; ++i) {
      const int u = tid + i * 256, wrow = u >> 3, k8 = u & 7;
      const int grow = (wrow >> 5) * HID + nBase + (wrow & 31);
      pW[i] = *(const u32x4*)(wS + (size_t)grow * DIM + kOff + k8 * 8);
    }
  };
  auto store_stage = [&](int idx) {
    unsigned short* dA = smem + idx * BUF_SH;
    unsigned short* dW = smem + idx * BUF_SH + A_SH;
    #pragma unroll
    for (int i = 0; i < 8; ++i) {
      const int u = tid + i * 256, row = u >> 3, k8 = u & 7;
      *(u32x4*)&dA[row * LDA + k8 * 8] = pA[i];
    }
    #pragma unroll
    for (int i = 0; i < 4; ++i) {
      const int u = tid + i * 256, wrow = u >> 3, k8 = u & 7;
      *(u32x4*)&dW[wrow * LDA + k8 * 8] = pW[i];
    }
  };
  load_regs(0);
  store_stage(0);
  for (int kb = 0; kb < KTOT; kb += BK) {
    const int idx = (kb >> 6) & 1;
    const bool more = kb + BK < KTOT;
    if (more) load_regs(kb + BK);
    __syncthreads();
    if (more) store_stage(idx ^ 1);
    wmma_stage<2>(smem + idx * BUF_SH, smem + idx * BUF_SH + A_SH,
                  acc, wave, nlo, half);
  }
#endif

  lstm_epilogue<2>(acc, c0, b_ih, b_hh, out, rowBase, nBase, wave, nlo, half);
}

// ---------------- small-ws fallback: fused f32-staging kernel ---------------
__global__ __launch_bounds__(256)
void lstm_fused_wmma_f32(const float* __restrict__ x, const float* __restrict__ h0,
                         const float* __restrict__ c0, const float* __restrict__ w_ih,
                         const float* __restrict__ w_hh, const float* __restrict__ b_ih,
                         const float* __restrict__ b_hh, float* __restrict__ out)
{
  __shared__ __align__(16) unsigned short sA[128 * LDA];
  __shared__ __align__(16) unsigned short sW[128 * LDA];
  const int tid = threadIdx.x, wave = tid >> 5, lane = tid & 31;
  const int nlo = lane & 15, half = lane >> 4;
  const int rowBase = blockIdx.y * 128, nBase = blockIdx.x * BN;

  v8f acc[4][2][1];
  { v8f z = {};
    for (int g = 0; g < 4; ++g) for (int t = 0; t < 2; ++t) acc[g][t][0] = z; }

  for (int kb = 0; kb < KTOT; kb += BK) {
    const float* aSrc = (kb < DIM) ? x : h0;
    const float* wSrc = (kb < DIM) ? w_ih : w_hh;
    const int kOff = kb & (DIM - 1);
    #pragma unroll
    for (int i = 0; i < 8; ++i) {
      const int u = tid + i * 256, row = u >> 4, k4 = u & 15;
      f32x4 v = *(const f32x4*)(aSrc + (size_t)(rowBase + row) * DIM + kOff + k4 * 4);
      u32x2 p; p.x = f2bf(v.x) | (f2bf(v.y) << 16); p.y = f2bf(v.z) | (f2bf(v.w) << 16);
      *(u32x2*)&sA[row * LDA + k4 * 4] = p;
    }
    #pragma unroll
    for (int i = 0; i < 8; ++i) {
      const int u = tid + i * 256, wrow = u >> 4, k4 = u & 15;
      const int grow = (wrow >> 5) * HID + nBase + (wrow & 31);
      f32x4 v = *(const f32x4*)(wSrc + (size_t)grow * DIM + kOff + k4 * 4);
      u32x2 p; p.x = f2bf(v.x) | (f2bf(v.y) << 16); p.y = f2bf(v.z) | (f2bf(v.w) << 16);
      *(u32x2*)&sW[wrow * LDA + k4 * 4] = p;
    }
    __syncthreads();
    wmma_stage<1>(sA, sW, acc, wave, nlo, half);
    __syncthreads();
  }
  lstm_epilogue<1>(acc, c0, b_ih, b_hh, out, rowBase, nBase, wave, nlo, half);
}

extern "C" void kernel_launch(void* const* d_in, const int* in_sizes, int n_in,
                              void* d_out, int out_size, void* d_ws, size_t ws_size,
                              hipStream_t stream) {
  const float* x    = (const float*)d_in[0];
  const float* h0   = (const float*)d_in[1];
  const float* c0   = (const float*)d_in[2];
  const float* w_ih = (const float*)d_in[3];
  const float* w_hh = (const float*)d_in[4];
  const float* b_ih = (const float*)d_in[5];
  const float* b_hh = (const float*)d_in[6];
  float* out = (float*)d_out;

  const size_t need = (size_t)4 * NELEM * sizeof(unsigned short);  // 32 MB
  if (ws_size >= need) {
    unsigned short* ws = (unsigned short*)d_ws;
    unsigned short* xb   = ws + 0 * (size_t)NELEM;
    unsigned short* hb   = ws + 1 * (size_t)NELEM;
    unsigned short* wihb = ws + 2 * (size_t)NELEM;
    unsigned short* whhb = ws + 3 * (size_t)NELEM;

    dim3 cgrid(NELEM / (256 * 8), 4);
    convert_to_bf16<<<cgrid, 256, 0, stream>>>(x, h0, w_ih, w_hh, ws);

    (void)hipFuncSetAttribute((const void*)lstm_gemm_bf16,
                              hipFuncAttributeMaxDynamicSharedMemorySize,
                              SMEM_BYTES);
    dim3 grid(HID / BN, B_ROWS / BM);   // (32, 16)
    lstm_gemm_bf16<<<grid, 256, SMEM_BYTES, stream>>>(xb, hb, wihb, whhb,
                                                      c0, b_ih, b_hh, out);
  } else {
    dim3 grid(HID / BN, B_ROWS / 128);  // (32, 32)
    lstm_fused_wmma_f32<<<grid, 256, 0, stream>>>(x, h0, c0, w_ih, w_hh,
                                                  b_ih, b_hh, out);
  }
}
